// LiquidODECell_23270132809825
// MI455X (gfx1250) — compile-verified, matrix operand order checked by reading, not run on hardware
//
#include <hip/hip_runtime.h>
#include <hip/hip_bf16.h>
#include <cstdint>
#include <cstddef>

// ---------------------------------------------------------------------------
// Liquid ODE cell for MI455X (gfx1250):
//  - all GEMMs on v_wmma_f32_16x16x32_bf16 (f32 accumulate)
//  - LDS tile staging via GLOBAL_LOAD_ASYNC_TO_LDS_B128 (ASYNCcnt path),
//    overlapped with WMMA compute of the other buffer
//  - pointwise math (tanh / silu / softplus / RK2 update) fused into epilogues
//  wave32, 8 waves / 256-thread block, 128x128 output tile, K-step 32.
// ---------------------------------------------------------------------------

typedef __attribute__((ext_vector_type(16))) __bf16 v16bf;
typedef __attribute__((ext_vector_type(8)))  float  v8f;

#define TILE_M     128
#define TILE_N     128
#define TILE_K     32
#define LDS_STRIDE 40     // 80B row pitch: 16B aligned, staggers banks
#define OUT_N      2048   // all outputs are [4096, 2048]

__device__ __forceinline__ __bf16 f2bf(float f) {
  union { float f; uint32_t u; } a; a.f = f;
  uint32_t u = a.u;
  uint16_t r = (uint16_t)((u + 0x7FFFu + ((u >> 16) & 1u)) >> 16);  // RNE
  union { uint16_t s; __bf16 b; } o; o.s = r; return o.b;
}

// Stage a 128x32 bf16 tile (global, row-major, leading dim ld) into LDS using
// async global->LDS copies (no VGPR staging, tracked by ASYNCcnt).
// 512 16-byte chunks / 256 threads = 2 per thread; fully uniform.
__device__ __forceinline__ void stage_async(const __bf16* __restrict__ g, int ld,
                                            __bf16 (*lds)[LDS_STRIDE],
                                            int tid, int k0) {
#pragma unroll
  for (int i = 0; i < 2; ++i) {
    int id = tid + (i << 8);
    int r  = id >> 2;
    int c  = id & 3;
    unsigned long long src = (unsigned long long)(g + (size_t)r * ld + k0 + c * 8);
    uint32_t dst = (uint32_t)(uintptr_t)(&lds[r][c * 8]);  // DS byte offset
    asm volatile("global_load_async_to_lds_b128 %0, %1, off"
                 :
                 : "v"(dst), "v"(src)
                 : "memory");
  }
}

__device__ __forceinline__ void wait_async0() {
  asm volatile("s_wait_asynccnt 0x0" ::: "memory");
}

union Frag { v16bf v; uint4 q[2]; };

// C = act(A[M,K] @ W[N,K]^T + bias) with fused epilogues.
// MODE 0: out = s + b                       -> outF   (ih projection)
// MODE 1: out = tanh(s + ep0 + b)           -> outF   (interaction; ep0 = ih)
// MODE 2: out = silu(s + b)                 -> outB   (t1, bf16)
// MODE 3: tau = softplus(s + b) + 0.2
//         out = ep2 + coef*(ep0 - ep1)/tau  -> outF   (RK2 update)
template <int MODE>
__global__ __launch_bounds__(256) void gemm_bf16_wmma(
    const __bf16* __restrict__ A, int lda,
    const __bf16* __restrict__ W, int ldw, int K,
    const float*  __restrict__ bias,
    const float*  __restrict__ ep0,
    const float*  __restrict__ ep1,
    const float*  __restrict__ ep2,
    float coef,
    float*  __restrict__ outF,
    __bf16* __restrict__ outB) {
  __shared__ __align__(16) __bf16 ldsA[2][TILE_M][LDS_STRIDE];
  __shared__ __align__(16) __bf16 ldsB[2][TILE_N][LDS_STRIDE];

  const int tid  = threadIdx.x;
  const int lane = tid & 31;
  const int half = lane >> 4;   // 0: lanes 0-15, 1: lanes 16-31
  const int lr   = lane & 15;
  const int w    = tid >> 5;    // wave id 0..7
  const int waveM = w >> 2;     // 0..1  (64 rows each)
  const int waveN = w & 3;      // 0..3  (32 cols each)

  const __bf16* Ab = A + (size_t)(blockIdx.x * TILE_M) * lda;
  const __bf16* Wb = W + (size_t)(blockIdx.y * TILE_N) * ldw;
  const int nk = K / TILE_K;

  v8f acc[4][2];
  const v8f vzero = {0.f, 0.f, 0.f, 0.f, 0.f, 0.f, 0.f, 0.f};
#pragma unroll
  for (int mf = 0; mf < 4; ++mf)
#pragma unroll
    for (int nf = 0; nf < 2; ++nf) acc[mf][nf] = vzero;

  stage_async(Ab, lda, ldsA[0], tid, 0);
  stage_async(Wb, ldw, ldsB[0], tid, 0);
  wait_async0();
  __syncthreads();

  for (int kt = 0; kt < nk; ++kt) {
    const int cur = kt & 1;
    if (kt + 1 < nk) {  // async-fill the other buffer while we compute
      stage_async(Ab, lda, ldsA[cur ^ 1], tid, (kt + 1) * TILE_K);
      stage_async(Wb, ldw, ldsB[cur ^ 1], tid, (kt + 1) * TILE_K);
    }

    // A fragment (16-bit A 16x32 layout): lanes 0-15 hold K 0..7 & 16..23,
    // lanes 16-31 hold K 8..15 & 24..31, row M = lr. B mirrors with N = lr.
    Frag fa[4], fb[2];
#pragma unroll
    for (int mf = 0; mf < 4; ++mf) {
      const __bf16* p = &ldsA[cur][waveM * 64 + mf * 16 + lr][half * 8];
      fa[mf].q[0] = *reinterpret_cast<const uint4*>(p);
      fa[mf].q[1] = *reinterpret_cast<const uint4*>(p + 16);
    }
#pragma unroll
    for (int nf = 0; nf < 2; ++nf) {
      const __bf16* p = &ldsB[cur][waveN * 32 + nf * 16 + lr][half * 8];
      fb[nf].q[0] = *reinterpret_cast<const uint4*>(p);
      fb[nf].q[1] = *reinterpret_cast<const uint4*>(p + 16);
    }

#pragma unroll
    for (int mf = 0; mf < 4; ++mf)
#pragma unroll
      for (int nf = 0; nf < 2; ++nf)
        acc[mf][nf] = __builtin_amdgcn_wmma_f32_16x16x32_bf16(
            false, fa[mf].v, false, fb[nf].v, (short)0, acc[mf][nf],
            false, false);

    if (kt + 1 < nk) wait_async0();  // next tile fully in LDS
    __syncthreads();
  }

  // Epilogue. f32 C/D layout: VGPR r, lanes 0-15 -> (M=r, N=lane);
  // lanes 16-31 -> (M=8+r, N=lane-16).
  const int rowBase = blockIdx.x * TILE_M + waveM * 64;
  const int colBase = blockIdx.y * TILE_N + waveN * 32;
#pragma unroll
  for (int mf = 0; mf < 4; ++mf) {
#pragma unroll
    for (int nf = 0; nf < 2; ++nf) {
#pragma unroll
      for (int r = 0; r < 8; ++r) {
        const int row = rowBase + mf * 16 + half * 8 + r;
        const int col = colBase + nf * 16 + lr;
        const size_t idx = (size_t)row * OUT_N + col;
        const float s = acc[mf][nf][r];
        if (MODE == 0) {
          outF[idx] = s + bias[col];
        } else if (MODE == 1) {
          outF[idx] = tanhf(s + ep0[idx] + bias[col]);
        } else if (MODE == 2) {
          const float t = s + bias[col];
          outB[idx] = f2bf(t / (1.f + expf(-t)));
        } else {  // MODE == 3
          const float t  = s + bias[col];
          const float sp = (t > 0.f) ? (t + log1pf(expf(-t))) : log1pf(expf(t));
          const float tau = sp + 0.2f;                 // TAU_MIN
          const float dh  = (ep0[idx] - ep1[idx]) / tau;
          outF[idx] = ep2[idx] + coef * dh;
        }
      }
    }
  }
}

// float -> bf16 with independent source/dest leading dims (used to pack the
// h-columns of z = [x | h] without a copy).
__global__ void cvt_f32_bf16(const float* __restrict__ src, __bf16* __restrict__ dst,
                             long total, int cols, int sld, int dld) {
  long i = (long)blockIdx.x * blockDim.x + threadIdx.x;
  if (i >= total) return;
  int r = (int)(i / cols);
  int c = (int)(i % cols);
  dst[(size_t)r * dld + c] = f2bf(src[(size_t)r * sld + c]);
}

extern "C" void kernel_launch(void* const* d_in, const int* in_sizes, int n_in,
                              void* d_out, int out_size, void* d_ws, size_t ws_size,
                              hipStream_t stream) {
  (void)in_sizes; (void)n_in; (void)out_size; (void)ws_size;
  constexpr int B = 4096, D_IN = 1024, H = 2048, Z = D_IN + H;

  const float* x    = (const float*)d_in[0];
  const float* h    = (const float*)d_in[1];
  const float* W_ih = (const float*)d_in[2];
  const float* b_ih = (const float*)d_in[3];
  const float* W_hh = (const float*)d_in[4];
  const float* b_hh = (const float*)d_in[5];
  const float* W_t1 = (const float*)d_in[6];
  const float* b_t1 = (const float*)d_in[7];
  const float* W_t2 = (const float*)d_in[8];
  const float* b_t2 = (const float*)d_in[9];

  char* ws = (char*)d_ws;
  auto take = [&](size_t bytes) {
    char* p = ws;
    ws += (bytes + 255) & ~(size_t)255;
    return p;
  };
  __bf16* z    = (__bf16*)take((size_t)B * Z * 2);       // [x | h] in bf16
  __bf16* wih  = (__bf16*)take((size_t)H * D_IN * 2);
  __bf16* whh  = (__bf16*)take((size_t)H * H * 2);
  __bf16* wt1  = (__bf16*)take((size_t)H * Z * 2);
  __bf16* wt2  = (__bf16*)take((size_t)H * H * 2);
  __bf16* t1bf = (__bf16*)take((size_t)B * H * 2);
  float*  ihf   = (float*)take((size_t)B * H * 4);
  float*  interf= (float*)take((size_t)B * H * 4);
  float*  hcur  = (float*)take((size_t)B * H * 4);
  float*  hmid  = (float*)take((size_t)B * H * 4);

  auto cvt = [&](const float* s, __bf16* d, int rows, int cols, int sld, int dld) {
    long total = (long)rows * cols;
    cvt_f32_bf16<<<(int)((total + 255) / 256), 256, 0, stream>>>(s, d, total, cols, sld, dld);
  };

  // One-time packing: weights + x (x lives in z cols [0,1024) with ld=3072).
  cvt(x,    z,   B, D_IN, D_IN, Z);
  cvt(W_ih, wih, H, D_IN, D_IN, D_IN);
  cvt(W_hh, whh, H, H,    H,    H);
  cvt(W_t1, wt1, H, Z,    Z,    Z);
  cvt(W_t2, wt2, H, H,    H,    H);
  hipMemcpyAsync(hcur, h, (size_t)B * H * 4, hipMemcpyDeviceToDevice, stream);

  const dim3 grid(B / TILE_M, H / TILE_N);  // 32 x 16 blocks

  // ih = x @ W_ih^T + b_ih  (constant across ODE steps)
  gemm_bf16_wmma<0><<<grid, 256, 0, stream>>>(z, Z, wih, D_IN, D_IN,
                                              b_ih, nullptr, nullptr, nullptr,
                                              0.f, ihf, nullptr);

  const float dt = 1.0f / 3.0f;
  for (int step = 0; step < 3; ++step) {
    for (int sub = 0; sub < 2; ++sub) {   // sub 0: k1 -> h_mid, sub 1: k2 -> h
      const float* hin = sub ? hmid : hcur;
      // pack h_in into z cols [1024, 3072)
      cvt(hin, z + D_IN, B, H, H, Z);
      // interaction = tanh(ih + h_in @ W_hh^T + b_hh)
      gemm_bf16_wmma<1><<<grid, 256, 0, stream>>>(z + D_IN, Z, whh, H, H,
                                                  b_hh, ihf, nullptr, nullptr,
                                                  0.f, interf, nullptr);
      // t1 = silu([x|h_in] @ W_t1^T + b_t1)   (bf16 output feeds next GEMM)
      gemm_bf16_wmma<2><<<grid, 256, 0, stream>>>(z, Z, wt1, Z, Z,
                                                  b_t1, nullptr, nullptr, nullptr,
                                                  0.f, nullptr, t1bf);
      // tau = softplus(t1 @ W_t2^T + b_t2) + 0.2 ; RK2 update fused
      const float coef = sub ? dt : 0.5f * dt;
      float* outp = sub ? ((step == 2) ? (float*)d_out : hcur) : hmid;
      gemm_bf16_wmma<3><<<grid, 256, 0, stream>>>(t1bf, H, wt2, H, H,
                                                  b_t2, interf, hin, hcur,
                                                  coef, outp, nullptr);
    }
  }
}
